// Indexer_21148418966045
// MI455X (gfx1250) — matrix operand order repaired and checked by reading, not compiled, not run on hardware
//
#include <hip/hip_runtime.h>
#include <hip/hip_bf16.h>
#include <stdint.h>

typedef __attribute__((ext_vector_type(16))) _Float16 v16h;
typedef __attribute__((ext_vector_type(8)))  _Float16 v8h;
typedef __attribute__((ext_vector_type(4)))  _Float16 v4h;
typedef __attribute__((ext_vector_type(8)))  float    v8f;

#define SEQ    2048
#define DIM    2048
#define QLR    512
#define NH     16
#define HD     64
#define NQ     (NH*HD)       // 1024
#define NKW    80            // 64 k-dims + 16 head-weights, packed GEMM N
#define TOPKK  512
#define MASK_VAL (-2.3819763e38f)   // -0.7 * FLT_MAX
#define WSCALE (0.03125f)           // 16^-0.5 * 64^-0.5

union AFrag { v16h v; v8h h[2]; };

// ------------------------------------------------------------ fp32 -> f16 (x4)
__global__ void cvt4_kernel(const float* __restrict__ in, _Float16* __restrict__ out) {
    int i = (blockIdx.x * 256 + threadIdx.x) * 4;
    float4 v = *(const float4*)(in + i);
    v4h o = { (_Float16)v.x, (_Float16)v.y, (_Float16)v.z, (_Float16)v.w };
    *(v4h*)(out + i) = o;
}

// wq_b [QLR][NQ] fp32 -> wqb_t [NQ][QLR] f16 (transpose so B frags are contiguous)
__global__ void cvt_wqb_kernel(const float* __restrict__ w, _Float16* __restrict__ out) {
    int i = blockIdx.x * 256 + threadIdx.x;           // NQ*QLR out-indices
    int n = i >> 9, r = i & 511;
    out[i] = (_Float16)w[(size_t)r * NQ + n];
}

// pack [wk | wp] transposed: wcat_t [NKW][DIM] f16; n<64 -> wk[:,n], else wp[:,n-64]
__global__ void cvt_wcat_kernel(const float* __restrict__ wk, const float* __restrict__ wp,
                                _Float16* __restrict__ out) {
    int i = blockIdx.x * 256 + threadIdx.x;           // NKW*DIM out-indices
    int n = i / DIM, e = i % DIM;
    float v = (n < HD) ? wk[(size_t)e * HD + n] : wp[(size_t)e * NH + (n - HD)];
    out[i] = (_Float16)v;
}

// ---------------------------------------- [kraw|wraw] = x @ [wk|wp] (WMMA f16)
__global__ void kgemm_kernel(const _Float16* __restrict__ x_h,
                             const _Float16* __restrict__ wcat_t,
                             float* __restrict__ kw_raw) {
    const int tid = threadIdx.x, lane = tid & 31, wid = tid >> 5; // block = 160 (5 waves)
    const int n_base = wid * 16;                                   // 5 n-tiles cover N=80
    const int m_base = blockIdx.x * 16;
    const int m = lane & 15, kh = lane >> 4;
    v8f c = {};
    for (int kb = 0; kb < DIM / 32; ++kb) {
        AFrag a;
        const _Float16* ap = x_h + (size_t)(m_base + m) * DIM + kb * 32 + kh * 16;
        a.h[0] = *(const v8h*)ap;
        a.h[1] = *(const v8h*)(ap + 8);
        const _Float16* bp = wcat_t + (size_t)(n_base + m) * DIM + kb * 32 + kh * 16;
        v16h b = *(const v16h*)bp;
        c = __builtin_amdgcn_wmma_f32_16x16x32_f16(false, a.v, false, b, (short)0, c, false, false);
    }
    for (int i = 0; i < 8; ++i)
        kw_raw[(size_t)(m_base + i + 8 * kh) * NKW + n_base + m] = c[i];
}

// ------------------- per-row epilogue: LN + rope -> k_f16, scale -> weights f32
__global__ void kpost_kernel(const float* __restrict__ kw_raw,
                             const int*   __restrict__ positions,
                             const float* __restrict__ ln_s,
                             const float* __restrict__ ln_b,
                             _Float16* __restrict__ k_h,
                             float* __restrict__ w_out) {
    int t = blockIdx.x * 256 + threadIdx.x;           // one thread per row, 2048 rows
    if (t >= SEQ) return;
    const float* kr = kw_raw + (size_t)t * NKW;
    float mu = 0.f;
    for (int i = 0; i < HD; ++i) mu += kr[i];
    mu *= (1.f / HD);
    float var = 0.f;
    for (int i = 0; i < HD; ++i) { float d = kr[i] - mu; var += d * d; }
    float rsig = rsqrtf(var * (1.f / HD) + 1e-6f);
    float kn[HD];
    for (int i = 0; i < HD; ++i) kn[i] = (kr[i] - mu) * rsig * ln_s[i] + ln_b[i];
    float pos = (float)positions[t];
    _Float16* ko = k_h + (size_t)t * HD;
    for (int i = 0; i < 16; ++i) {                    // rope pairs (i, i+16)
        float invf = __powf(10000.f, -(float)i * (1.f / 16.f));
        float ang = pos * invf;
        float c = __cosf(ang), s = __sinf(ang);
        ko[i]      = (_Float16)(kn[i] * c - kn[i + 16] * s);
        ko[i + 16] = (_Float16)(kn[i + 16] * c + kn[i] * s);
    }
    for (int i = 32; i < HD; ++i) ko[i] = (_Float16)kn[i];
    for (int h = 0; h < NH; ++h) w_out[t * NH + h] = kr[HD + h] * WSCALE;
}

// ------------------------------------------------- q = qr @ wq_b   (WMMA f16)
__global__ void qgemm_kernel(const _Float16* __restrict__ qr_h,
                             const _Float16* __restrict__ wqb_t,
                             _Float16* __restrict__ q_h) {
    const int tid = threadIdx.x, lane = tid & 31, wid = tid >> 5; // block = 128, 4 waves
    const int n_base = (blockIdx.x * 4 + wid) * 16;
    const int s_base = blockIdx.y * 16;
    const int m = lane & 15, kh = lane >> 4;
    v8f c = {};
    for (int kb = 0; kb < QLR / 32; ++kb) {
        AFrag a;
        const _Float16* ap = qr_h + (size_t)(s_base + m) * QLR + kb * 32 + kh * 16;
        a.h[0] = *(const v8h*)ap;
        a.h[1] = *(const v8h*)(ap + 8);
        const _Float16* bp = wqb_t + (size_t)(n_base + m) * QLR + kb * 32 + kh * 16;
        v16h b = *(const v16h*)bp;
        c = __builtin_amdgcn_wmma_f32_16x16x32_f16(false, a.v, false, b, (short)0, c, false, false);
    }
    for (int i = 0; i < 8; ++i)
        q_h[(size_t)(s_base + i + 8 * kh) * NQ + n_base + m] = (_Float16)c[i];
}

// ------------------------------------------------- in-place partial rope on q
__global__ void qrope_kernel(_Float16* __restrict__ q_h, const int* __restrict__ positions) {
    int idx = blockIdx.x * 256 + threadIdx.x;         // SEQ*NH items
    int s = idx >> 4, h = idx & 15;
    _Float16* qp = q_h + (size_t)s * NQ + h * HD;
    float pos = (float)positions[s];
    for (int i = 0; i < 16; ++i) {
        float invf = __powf(10000.f, -(float)i * (1.f / 16.f));
        float ang = pos * invf;
        float c = __cosf(ang), sn = __sinf(ang);
        float x1 = (float)qp[i], x2 = (float)qp[i + 16];
        qp[i]      = (_Float16)(x1 * c - x2 * sn);
        qp[i + 16] = (_Float16)(x2 * c + x1 * sn);
    }
}

// ---------------- score[s,t] = sum_h relu(q[s,h]·k[t]) * w[s,h]   (WMMA f16)
__global__ void __launch_bounds__(256)
logits_score_kernel(const _Float16* __restrict__ q_h,
                    const _Float16* __restrict__ k_h,
                    const float*    __restrict__ w,
                    float* __restrict__ score) {
    __shared__ __align__(16) _Float16 qtile[16 * NQ]; // 32 KB: 16 query rows, all heads
    __shared__ float wtile[16 * NH];                  // 1 KB: weights for this s-tile
    const int tid = threadIdx.x, lane = tid & 31, wid = tid >> 5;
    const int s_base = blockIdx.y * 16;
    const int t_base = (blockIdx.x * 8 + wid) * 16;
    // stage 16 full q rows (contiguous 32KB) + 16x16 weight tile into LDS
    {
        const uint4* src = (const uint4*)(q_h + (size_t)s_base * NQ);
        uint4* dst = (uint4*)qtile;
        for (int i = tid; i < 2048; i += 256) dst[i] = src[i];
        wtile[tid] = w[s_base * NH + tid];            // 256 floats, coalesced
    }
    __syncthreads();
    const int m = lane & 15, kh = lane >> 4;
    // B operand (k tile, K=d) lives in VGPRs for the whole head loop
    const _Float16* kp = k_h + (size_t)(t_base + m) * HD + kh * 16;
    v16h b0 = *(const v16h*)kp;          // d = 0..31  chunk
    v16h b1 = *(const v16h*)(kp + 32);   // d = 32..63 chunk
    v8f acc = {};
    for (int h = 0; h < NH; ++h) {
        AFrag a0, a1;
        const _Float16* qp = qtile + m * NQ + h * HD + kh * 16;
        a0.h[0] = *(const v8h*)qp;
        a0.h[1] = *(const v8h*)(qp + 8);
        a1.h[0] = *(const v8h*)(qp + 32);
        a1.h[1] = *(const v8h*)(qp + 40);
        v8f c = {};
        c = __builtin_amdgcn_wmma_f32_16x16x32_f16(false, a0.v, false, b0, (short)0, c, false, false);
        c = __builtin_amdgcn_wmma_f32_16x16x32_f16(false, a1.v, false, b1, (short)0, c, false, false);
        for (int i = 0; i < 8; ++i) {
            float wv = wtile[(i + 8 * kh) * NH + h];
            float l = c[i];
            l = l > 0.f ? l : 0.f;       // relu
            acc[i] += l * wv;
        }
    }
    for (int i = 0; i < 8; ++i)
        score[(size_t)(s_base + i + 8 * kh) * SEQ + t_base + m] = acc[i];
}

// ---------------------------------------------- exact top-512 + additive mask
__device__ __forceinline__ unsigned f2ord(float f) {
    unsigned u = __float_as_uint(f);
    return (u & 0x80000000u) ? ~u : (u | 0x80000000u);
}

__global__ void __launch_bounds__(256)
topk_mask_kernel(const float* __restrict__ score,
                 float* __restrict__ mask,
                 int*   __restrict__ idx_out) {
    __shared__ unsigned keys[SEQ];
    __shared__ int gcnt[256], ecnt[256], gpre[256], epre[256];
    __shared__ int scnt, gtot;
    const int row = blockIdx.x, tid = threadIdx.x;
    const float* sr = score + (size_t)row * SEQ;
    float* mr = mask + (size_t)row * SEQ;
    int* ir = idx_out + (size_t)row * TOPKK;
    for (int i = tid; i < SEQ; i += 256) { keys[i] = f2ord(sr[i]); mr[i] = MASK_VAL; }
    __syncthreads();
    // binary search: largest thr with count(keys >= thr) >= TOPKK
    unsigned thr = 0u;
    const int base8 = tid * 8;
    for (int bit = 31; bit >= 0; --bit) {
        unsigned cand = thr | (1u << bit);
        if (tid == 0) scnt = 0;
        __syncthreads();
        int c = 0;
        for (int i = 0; i < 8; ++i) c += (keys[base8 + i] >= cand) ? 1 : 0;
        atomicAdd(&scnt, c);
        __syncthreads();
        if (scnt >= TOPKK) thr = cand;
        __syncthreads();
    }
    int g = 0, e = 0;
    for (int i = 0; i < 8; ++i) {
        unsigned k = keys[base8 + i];
        g += (k > thr) ? 1 : 0;
        e += (k == thr) ? 1 : 0;
    }
    gcnt[tid] = g; ecnt[tid] = e;
    __syncthreads();
    if (tid == 0) {
        int sg = 0, se = 0;
        for (int i = 0; i < 256; ++i) {
            gpre[i] = sg; sg += gcnt[i];
            epre[i] = se; se += ecnt[i];
        }
        gtot = sg;
    }
    __syncthreads();
    int gp = gpre[tid];
    int ep = gtot + epre[tid];
    for (int i = 0; i < 8; ++i) {
        int idx = base8 + i;
        unsigned k = keys[idx];
        if (k > thr) {
            ir[gp++] = idx; mr[idx] = 0.f;
        } else if (k == thr && ep < TOPKK) {
            ir[ep++] = idx; mr[idx] = 0.f;
        }
    }
}

extern "C" void kernel_launch(void* const* d_in, const int* in_sizes, int n_in,
                              void* d_out, int out_size, void* d_ws, size_t ws_size,
                              hipStream_t stream) {
    (void)in_sizes; (void)n_in; (void)out_size; (void)ws_size;
    const float* x    = (const float*)d_in[0];
    const float* qr   = (const float*)d_in[1];
    const int*   pos  = (const int*)  d_in[2];
    const float* wq_b = (const float*)d_in[3];
    const float* wk   = (const float*)d_in[4];
    const float* wp   = (const float*)d_in[5];
    const float* ln_s = (const float*)d_in[6];
    const float* ln_b = (const float*)d_in[7];

    // output layout (tuple, flat): mask[1,1,S,S] f32 | topk_indices[1,S,512] i32 | index_score[1,S,S] f32
    float* mask  = (float*)d_out;
    int*   tidx  = (int*)((float*)d_out + (size_t)SEQ * SEQ);
    float* score = (float*)d_out + (size_t)SEQ * SEQ + (size_t)SEQ * TOPKK;

    // workspace layout
    char* ws = (char*)d_ws;
    _Float16* x_h    = (_Float16*)(ws);                          // 8 MB
    _Float16* qr_h   = (_Float16*)(ws + 8388608);                // 2 MB
    _Float16* wqb_t  = (_Float16*)(ws + 10485760);               // 1 MB
    _Float16* wcat_t = (_Float16*)(ws + 11534336);               // 320 KB
    _Float16* q_h    = (_Float16*)(ws + 11862016);               // 4 MB
    float*    kw_raw = (float*)   (ws + 16056320);               // 640 KB
    _Float16* k_h    = (_Float16*)(ws + 16711680);               // 256 KB
    float*    wbuf   = (float*)   (ws + 16973824);               // 128 KB

    cvt4_kernel    <<<(SEQ * DIM) / 1024, 256, 0, stream>>>(x, x_h);
    cvt4_kernel    <<<(SEQ * QLR) / 1024, 256, 0, stream>>>(qr, qr_h);
    cvt_wqb_kernel <<<(NQ * QLR) / 256, 256, 0, stream>>>(wq_b, wqb_t);
    cvt_wcat_kernel<<<(NKW * DIM) / 256, 256, 0, stream>>>(wk, wp, wcat_t);
    kgemm_kernel   <<<SEQ / 16, 160, 0, stream>>>(x_h, wcat_t, kw_raw);
    kpost_kernel   <<<SEQ / 256, 256, 0, stream>>>(kw_raw, pos, ln_s, ln_b, k_h, wbuf);
    qgemm_kernel   <<<dim3(NQ / 64, SEQ / 16), 128, 0, stream>>>(qr_h, wqb_t, q_h);
    qrope_kernel   <<<(SEQ * NH) / 256, 256, 0, stream>>>(q_h, pos);
    logits_score_kernel<<<dim3(SEQ / 128, SEQ / 16), 256, 0, stream>>>(q_h, k_h, wbuf, score);
    topk_mask_kernel<<<SEQ, 256, 0, stream>>>(score, mask, tidx);
}
